// Rel_Attention_58145267253941
// MI455X (gfx1250) — matrix operand order repaired and checked by reading, not compiled, not run on hardware
//
#include <hip/hip_runtime.h>
#include <hip/hip_bf16.h>
#include <math.h>

typedef __attribute__((ext_vector_type(16))) __bf16 v16bf;
typedef __attribute__((ext_vector_type(8)))  __bf16 v8bf;
typedef __attribute__((ext_vector_type(8)))  float  v8f;

#define WIN   7
#define N49   49
#define HEADS 8
#define DIM   256
#define HD    32

// ---- WMMA wave32 fragment maps (CDNA5 ISA 7.12.2), bf16 16x16x32 ----
// A (16x32, MxK): two contiguous 8-elem runs at K = 8*half, K = 16+8*half
// B (32x16, KxN): col = lane&15, two contiguous 8-elem runs at K = 8*half, K = 16+8*half
// C/D (16x16 f32): col = lane&15, row = r + 8*half

__device__ __forceinline__ v16bf load_a_frag(const __bf16* rowp, int half) {
  v8bf lo = *(const v8bf*)(rowp + 8 * half);
  v8bf hi = *(const v8bf*)(rowp + 16 + 8 * half);
  return __builtin_shufflevector(lo, hi, 0, 1, 2, 3, 4, 5, 6, 7,
                                 8, 9, 10, 11, 12, 13, 14, 15);
}
__device__ __forceinline__ v16bf load_b_frag(const __bf16* colp, int half) {
  v8bf lo = *(const v8bf*)(colp + 8 * half);
  v8bf hi = *(const v8bf*)(colp + 16 + 8 * half);
  return __builtin_shufflevector(lo, hi, 0, 1, 2, 3, 4, 5, 6, 7,
                                 8, 9, 10, 11, 12, 13, 14, 15);
}

// ------------------------------------------------------------------
// Setup: pos2 table, gathered rel bias, its row-sums, resize coeffs
// ------------------------------------------------------------------
__global__ void setup_tables(const float* __restrict__ rel_table,
                             float* __restrict__ pos2,
                             float* __restrict__ relb,
                             float* __restrict__ rbs,
                             int* __restrict__ ri0, int* __restrict__ ri1,
                             float* __restrict__ rw) {
  __shared__ float pe[N49][HD];
  const int t = threadIdx.x;
  for (int idx = t; idx < N49 * HD; idx += 256) {
    int n = idx >> 5, d = idx & 31;
    int i = d >> 1;
    float div = __expf((float)(2 * i) * (-logf(10000.f) / (float)HD));
    float ang = (float)n * div;
    pe[n][d] = (d & 1) ? cosf(ang) : sinf(ang);
  }
  __syncthreads();
  for (int idx = t; idx < N49 * N49; idx += 256) {
    int n = idx / N49, m = idx % N49;
    float s = 0.f;
    for (int d = 0; d < HD; ++d) s += pe[n][d] * pe[m][d];
    pos2[idx] = s;
  }
  for (int idx = t; idx < HEADS * N49 * N49; idx += 256) {
    int h = idx / (N49 * N49);
    int rem = idx % (N49 * N49);
    int n = rem / N49, m = rem % N49;
    int yn = n / WIN, xn = n % WIN, ym = m / WIN, xm = m % WIN;
    int ridx = ((yn - ym) + WIN - 1) * (2 * WIN - 1) + ((xn - xm) + WIN - 1);
    relb[idx] = rel_table[ridx * HEADS + h];
  }
  for (int idx = t; idx < HEADS * N49; idx += 256) {
    int h = idx / N49, n = idx % N49;
    int yn = n / WIN, xn = n % WIN;
    float s = 0.f;
    for (int m = 0; m < N49; ++m) {
      int ym = m / WIN, xm = m % WIN;
      int ridx = ((yn - ym) + WIN - 1) * (2 * WIN - 1) + ((xn - xm) + WIN - 1);
      s += rel_table[ridx * HEADS + h];
    }
    rbs[idx] = s;
  }
  if (t < N49) {  // half-pixel bilinear coords: 32 -> 49 along last dim
    float xm = ((float)t + 0.5f) * ((float)HD / (float)N49) - 0.5f;
    float xl = floorf(xm);
    float w = xm - xl;
    int i0 = (int)xl, i1 = (int)xl + 1;
    i0 = i0 < 0 ? 0 : (i0 > HD - 1 ? HD - 1 : i0);
    i1 = i1 < 0 ? 0 : (i1 > HD - 1 ? HD - 1 : i1);
    ri0[t] = i0; ri1[t] = i1; rw[t] = w;
  }
}

// ------------------------------------------------------------------
// QKV GEMM: qkv[b,o,n] = sum_c W[o,c]*x[b,c,n] + bias[o]  (bf16 out)
// Block tile 64(o) x 64(n); 8 waves x 2 tiles (shared B frag);
// double-buffered LDS (1 barrier / K-step). K = 256 in steps of 32.
// ------------------------------------------------------------------
__global__ void qkv_gemm(const float* __restrict__ x,
                         const float* __restrict__ w,
                         const float* __restrict__ bias,
                         __bf16* __restrict__ qkv) {
  __shared__ alignas(16) __bf16 Ws[2][64][40];   // pitch 80B
  __shared__ alignas(16) __bf16 XsT[2][64][40];
  const int t = threadIdx.x;
  const int b = blockIdx.y;
  const int mBase = blockIdx.x * 64;
  const int lane = t & 31, w8 = t >> 5;
  const int half = lane >> 4, ll = lane & 15;
  const int tn = w8 & 3;               // column tile
  const int tmA = w8 >> 2;             // row tiles tmA and tmA+2
  // staging maps
  const int so = t >> 2, sc0 = (t & 3) * 8;  // Ws: 8 consecutive c, fixed o
  const int sn = t & 63, sk0 = (t >> 6) * 8; // XsT: 8 consecutive c, fixed n
  const float* wsrc = w + (size_t)(mBase + so) * DIM + sc0;
  const float* xsrc = x + (size_t)b * (DIM * N49) + (size_t)sk0 * N49 + sn;

  auto stage = [&](int bufi, int k0) {
    { const float* s = wsrc + k0;
      v8bf p;
#pragma unroll
      for (int jj = 0; jj < 8; ++jj) p[jj] = (__bf16)s[jj];
      *(v8bf*)&Ws[bufi][so][sc0] = p; }
    { v8bf p = {};
      if (sn < N49) {
        const float* s = xsrc + (size_t)k0 * N49;
#pragma unroll
        for (int jj = 0; jj < 8; ++jj) p[jj] = (__bf16)s[(size_t)jj * N49];
      }
      *(v8bf*)&XsT[bufi][sn][sk0] = p; }
  };

  v8f acc0 = {}, acc1 = {};
  stage(0, 0);
  __syncthreads();
  int buf = 0;
  for (int k0 = 0; k0 < DIM; k0 += 32) {
    if (k0 + 32 < DIM) stage(buf ^ 1, k0 + 32);
    v16bf bf = load_b_frag(&XsT[buf][tn * 16 + ll][0], half);
    v16bf a0 = load_a_frag(&Ws[buf][tmA * 16 + ll][0], half);
    v16bf a1 = load_a_frag(&Ws[buf][(tmA + 2) * 16 + ll][0], half);
    acc0 = __builtin_amdgcn_wmma_f32_16x16x32_bf16(false, a0, false, bf,
                                                   (short)0, acc0, false, false);
    acc1 = __builtin_amdgcn_wmma_f32_16x16x32_bf16(false, a1, false, bf,
                                                   (short)0, acc1, false, false);
    __syncthreads();
    buf ^= 1;
  }
  int n = tn * 16 + ll;
  if (n < N49) {
#pragma unroll
    for (int r = 0; r < 8; ++r) {
      int o0 = mBase + tmA * 16 + r + half * 8;
      int o1 = o0 + 32;
      qkv[((size_t)b * 768 + o0) * N49 + n] = (__bf16)(acc0[r] + bias[o0]);
      qkv[((size_t)b * 768 + o1) * N49 + n] = (__bf16)(acc1[r] + bias[o1]);
    }
  }
}

// ------------------------------------------------------------------
// Attention scores: attn[b,h,n,m] = scale*q.k + pos2 + relb + v_adj
// one block per (h,b); q/k/v head slices staged K(d)-major in LDS
// ------------------------------------------------------------------
__global__ void attn_scores(const __bf16* __restrict__ qkv,
                            const float* __restrict__ pos2,
                            const float* __restrict__ relb,
                            const float* __restrict__ rbs,
                            const int* __restrict__ ri0,
                            const int* __restrict__ ri1,
                            const float* __restrict__ rw,
                            float* __restrict__ attn) {
  __shared__ alignas(16) __bf16 qs[64][40];
  __shared__ alignas(16) __bf16 ks[64][40];
  __shared__ alignas(16) __bf16 vs[64][40];
  __shared__ float rbsS[N49], rwS[N49];
  __shared__ int i0S[N49], i1S[N49];
  const int t = threadIdx.x;
  const int h = blockIdx.x, b = blockIdx.y;
  const int lane = t & 31, w8 = t >> 5;
  const int half = lane >> 4, ll = lane & 15;
  const size_t qb = ((size_t)b * 768 + h * HD) * N49;
  const size_t kb = ((size_t)b * 768 + 256 + h * HD) * N49;
  const size_t vb = ((size_t)b * 768 + 512 + h * HD) * N49;
  { const int sn = t & 63, sd0 = (t >> 6) * 8;   // 8 consecutive d, fixed n
    v8bf pq = {}, pk = {}, pv = {};
    if (sn < N49) {
#pragma unroll
      for (int jj = 0; jj < 8; ++jj) {
        size_t off = (size_t)(sd0 + jj) * N49 + sn;
        pq[jj] = qkv[qb + off];
        pk[jj] = qkv[kb + off];
        pv[jj] = qkv[vb + off];
      }
    }
    *(v8bf*)&qs[sn][sd0] = pq;
    *(v8bf*)&ks[sn][sd0] = pk;
    *(v8bf*)&vs[sn][sd0] = pv; }
  if (t < N49) { rbsS[t] = rbs[h * N49 + t]; rwS[t] = rw[t]; i0S[t] = ri0[t]; i1S[t] = ri1[t]; }
  __syncthreads();
  const float scale = 0.17677669529663687f;  // 32^-0.5
  for (int tt = w8; tt < 16; tt += 8) {      // 16 tiles over 8 waves
    int tm = tt >> 2, tn = tt & 3;
    v16bf af = load_a_frag(&qs[tm * 16 + ll][0], half);
    v16bf bf = load_b_frag(&ks[tn * 16 + ll][0], half);
    v8f acc = {};
    acc = __builtin_amdgcn_wmma_f32_16x16x32_bf16(false, af, false, bf,
                                                  (short)0, acc, false, false);
    int m = tn * 16 + ll;
    if (m < N49) {
      float wl = rwS[m];
      int j0 = i0S[m], j1 = i1S[m];
#pragma unroll
      for (int r = 0; r < 8; ++r) {
        int n = tm * 16 + r + half * 8;
        if (n < N49) {
          float vadj = rbsS[n] * ((1.f - wl) * (float)vs[n][j0] +
                                  wl * (float)vs[n][j1]);
          float val = acc[r] * scale + pos2[n * N49 + m] +
                      relb[(h * N49 + n) * N49 + m] + vadj;
          attn[(((size_t)b * HEADS + h) * N49 + n) * N49 + m] = val;
        }
      }
    }
  }
}

// ------------------------------------------------------------------
// Tiled 32x32 transpose: src[R][C] -> dst[C][R]
// ------------------------------------------------------------------
__global__ void transpose_k(const float* __restrict__ src,
                            float* __restrict__ dst, int R, int C) {
  __shared__ float tile[32][33];
  const int tx = threadIdx.x & 31, ty = threadIdx.x >> 5;
  const int c0 = blockIdx.x * 32, r0 = blockIdx.y * 32;
  for (int j = 0; j < 32; j += 8) {
    int r = r0 + ty + j, c = c0 + tx;
    tile[ty + j][tx] = (r < R && c < C) ? src[(size_t)r * C + c] : 0.f;
  }
  __syncthreads();
  for (int j = 0; j < 32; j += 8) {
    int c = c0 + ty + j, r = r0 + tx;
    if (c < C && r < R) dst[(size_t)c * R + r] = tile[tx][ty + j];
  }
}

// ------------------------------------------------------------------
// Per-row (h,n,m): radix-select k-th largest over Bw windows, mask,
// exp(x/bias_p), normalize over Bw. Row is contiguous (attnT layout).
// ------------------------------------------------------------------
__global__ void topk_softmax(float* __restrict__ attnT,
                             const float* __restrict__ biasP,
                             int ktop, int bw) {
  __shared__ int redI[256];
  __shared__ float redF[256];
  const int t = threadIdx.x;
  float* row = attnT + (size_t)blockIdx.x * bw;
  float v4[4]; unsigned u4[4];
  for (int j = 0; j < 4; ++j) {
    int i = t * 4 + j;
    float f = (i < bw) ? row[i] : 0.f;
    v4[j] = f;
    unsigned bi = __float_as_uint(f);
    unsigned u = (bi & 0x80000000u) ? ~bi : (bi | 0x80000000u);
    u4[j] = (i < bw) ? u : 0u;
  }
  unsigned prefix = 0;
  for (int bit = 31; bit >= 0; --bit) {
    unsigned cand = prefix | (1u << bit);
    int c = 0;
    for (int j = 0; j < 4; ++j) c += (u4[j] >= cand) ? 1 : 0;
    redI[t] = c; __syncthreads();
    for (int s = 128; s > 0; s >>= 1) {
      if (t < s) redI[t] += redI[t + s];
      __syncthreads();
    }
    int total = redI[0]; __syncthreads();
    if (total >= ktop) prefix = cand;   // largest T with count(>=T) >= k
  }
  float bp = biasP[0];
  float e4[4]; float s = 0.f;
  for (int j = 0; j < 4; ++j) {
    float e = (u4[j] >= prefix) ? expf(v4[j] / bp) : 0.f;  // no max-shift, per ref
    e4[j] = e; s += e;
  }
  redF[t] = s; __syncthreads();
  for (int sr = 128; sr > 0; sr >>= 1) {
    if (t < sr) redF[t] += redF[t + sr];
    __syncthreads();
  }
  float tot = redF[0];
  for (int j = 0; j < 4; ++j) {
    int i = t * 4 + j;
    if (i < bw) row[i] = e4[j] / tot;
  }
}

// ------------------------------------------------------------------
// attn2 @ V per (b,h): (49x49)(49x32), K(m) padded to 64
// As: K(m)-major rows; VsT: K(m)-major per column(d)
// ------------------------------------------------------------------
__global__ void av_gemm(const float* __restrict__ attn2,
                        const __bf16* __restrict__ qkv,
                        __bf16* __restrict__ outmid) {
  __shared__ alignas(16) __bf16 As[64][72];    // pitch 144B
  __shared__ alignas(16) __bf16 VsT[32][72];
  const int t = threadIdx.x;
  const int h = blockIdx.x, b = blockIdx.y;
  const int lane = t & 31, w8 = t >> 5;
  const int half = lane >> 4, ll = lane & 15;
  { const int sn = t >> 2, mq = (t & 3) * 16;  // 16 consecutive m, fixed n
    v8bf p0 = {}, p1 = {};
    if (sn < N49) {
      const float* src = attn2 + (((size_t)b * HEADS + h) * N49 + sn) * N49;
      if (mq < 48) {
#pragma unroll
        for (int jj = 0; jj < 8; ++jj) p0[jj] = (__bf16)src[mq + jj];
#pragma unroll
        for (int jj = 0; jj < 8; ++jj) p1[jj] = (__bf16)src[mq + 8 + jj];
      } else {
        p0[0] = (__bf16)src[48];               // only m=48 valid
      }
    }
    *(v8bf*)&As[sn][mq] = p0;
    *(v8bf*)&As[sn][mq + 8] = p1; }
  { const int sd = t >> 3, m0 = (t & 7) * 8;   // 8 consecutive m, fixed d
    const size_t vb = ((size_t)b * 768 + 512 + h * HD) * N49;
    v8bf p = {};
    if (m0 + 8 <= N49) {
#pragma unroll
      for (int jj = 0; jj < 8; ++jj) p[jj] = qkv[vb + (size_t)sd * N49 + m0 + jj];
    } else if (m0 < N49) {                     // m0 == 48
      p[0] = qkv[vb + (size_t)sd * N49 + 48];
    }
    *(v8bf*)&VsT[sd][m0] = p; }
  __syncthreads();
  const int tm = w8 >> 1, tn = w8 & 1;
  v8f acc = {};
#pragma unroll
  for (int kk = 0; kk < 2; ++kk) {
    v16bf af = load_a_frag(&As[tm * 16 + ll][kk * 32], half);
    v16bf bf = load_b_frag(&VsT[tn * 16 + ll][kk * 32], half);
    acc = __builtin_amdgcn_wmma_f32_16x16x32_bf16(false, af, false, bf,
                                                  (short)0, acc, false, false);
  }
  const int d = tn * 16 + ll;
#pragma unroll
  for (int r = 0; r < 8; ++r) {
    int n = tm * 16 + r + half * 8;
    if (n < N49)
      outmid[((size_t)b * DIM + h * HD + d) * N49 + n] = (__bf16)acc[r];
  }
}

// ------------------------------------------------------------------
// Projection GEMM: out[b,o,n] = sum_c Pw[o,c]*mid[b,c,n] + Pb[o] (f32 out)
// Same structure as qkv_gemm (64x64 tile, 2 tiles/wave, double buffer)
// ------------------------------------------------------------------
__global__ void proj_gemm(const __bf16* __restrict__ outmid,
                          const float* __restrict__ w,
                          const float* __restrict__ bias,
                          float* __restrict__ out) {
  __shared__ alignas(16) __bf16 Ws[2][64][40];
  __shared__ alignas(16) __bf16 XsT[2][64][40];
  const int t = threadIdx.x;
  const int b = blockIdx.y;
  const int mBase = blockIdx.x * 64;
  const int lane = t & 31, w8 = t >> 5;
  const int half = lane >> 4, ll = lane & 15;
  const int tn = w8 & 3;
  const int tmA = w8 >> 2;
  const int so = t >> 2, sc0 = (t & 3) * 8;
  const int sn = t & 63, sk0 = (t >> 6) * 8;
  const float* wsrc = w + (size_t)(mBase + so) * DIM + sc0;
  const __bf16* xsrc = outmid + (size_t)b * (DIM * N49) + (size_t)sk0 * N49 + sn;

  auto stage = [&](int bufi, int k0) {
    { const float* s = wsrc + k0;
      v8bf p;
#pragma unroll
      for (int jj = 0; jj < 8; ++jj) p[jj] = (__bf16)s[jj];
      *(v8bf*)&Ws[bufi][so][sc0] = p; }
    { v8bf p = {};
      if (sn < N49) {
        const __bf16* s = xsrc + (size_t)k0 * N49;
#pragma unroll
        for (int jj = 0; jj < 8; ++jj) p[jj] = s[(size_t)jj * N49];
      }
      *(v8bf*)&XsT[bufi][sn][sk0] = p; }
  };

  v8f acc0 = {}, acc1 = {};
  stage(0, 0);
  __syncthreads();
  int buf = 0;
  for (int k0 = 0; k0 < DIM; k0 += 32) {
    if (k0 + 32 < DIM) stage(buf ^ 1, k0 + 32);
    v16bf bf = load_b_frag(&XsT[buf][tn * 16 + ll][0], half);
    v16bf a0 = load_a_frag(&Ws[buf][tmA * 16 + ll][0], half);
    v16bf a1 = load_a_frag(&Ws[buf][(tmA + 2) * 16 + ll][0], half);
    acc0 = __builtin_amdgcn_wmma_f32_16x16x32_bf16(false, a0, false, bf,
                                                   (short)0, acc0, false, false);
    acc1 = __builtin_amdgcn_wmma_f32_16x16x32_bf16(false, a1, false, bf,
                                                   (short)0, acc1, false, false);
    __syncthreads();
    buf ^= 1;
  }
  int n = tn * 16 + ll;
  if (n < N49) {
#pragma unroll
    for (int r = 0; r < 8; ++r) {
      int o0 = mBase + tmA * 16 + r + half * 8;
      int o1 = o0 + 32;
      out[((size_t)b * DIM + o0) * N49 + n] = acc0[r] + bias[o0];
      out[((size_t)b * DIM + o1) * N49 + n] = acc1[r] + bias[o1];
    }
  }
}

// ------------------------------------------------------------------
extern "C" void kernel_launch(void* const* d_in, const int* in_sizes, int n_in,
                              void* d_out, int out_size, void* d_ws, size_t ws_size,
                              hipStream_t stream) {
  const float* x         = (const float*)d_in[0];
  const float* qkv_w     = (const float*)d_in[1];
  const float* qkv_b     = (const float*)d_in[2];
  const float* proj_w    = (const float*)d_in[3];
  const float* proj_b    = (const float*)d_in[4];
  const float* rel_table = (const float*)d_in[5];
  const float* bias_p    = (const float*)d_in[6];
  float* out = (float*)d_out;

  const int Bw = in_sizes[0] / (DIM * N49);      // 1024 windows
  const int P  = HEADS * N49 * N49;              // 19208 (h,n,m) rows

  char* ws = (char*)d_ws;
  auto al = [](size_t v) { return (v + 255) & ~(size_t)255; };
  const size_t SLACK = 4096;                     // tail slack for padded reads
  size_t off = 0;
  __bf16* qkv   = (__bf16*)(ws + off); off = al(off + (size_t)Bw * 768 * N49 * 2 + SLACK);
  float* attn   = (float*)(ws + off);  off = al(off + (size_t)Bw * P * 4 + SLACK);
  float* attnT  = (float*)(ws + off);  off = al(off + (size_t)Bw * P * 4 + SLACK);
  __bf16* mid   = (__bf16*)(ws + off); off = al(off + (size_t)Bw * DIM * N49 * 2 + SLACK);
  float* pos2   = (float*)(ws + off);  off = al(off + (size_t)N49 * N49 * 4);
  float* relb   = (float*)(ws + off);  off = al(off + (size_t)HEADS * N49 * N49 * 4);
  float* rbs    = (float*)(ws + off);  off = al(off + (size_t)HEADS * N49 * 4);
  int*   ri0    = (int*)(ws + off);    off = al(off + (size_t)N49 * 4);
  int*   ri1    = (int*)(ws + off);    off = al(off + (size_t)N49 * 4);
  float* rwv    = (float*)(ws + off);  off = al(off + (size_t)N49 * 4);

  const int ktop = (int)((1.0 / (1.0 + exp(-0.8))) * (double)Bw);  // 706

  setup_tables<<<1, 256, 0, stream>>>(rel_table, pos2, relb, rbs, ri0, ri1, rwv);
  qkv_gemm<<<dim3(768 / 64, Bw), 256, 0, stream>>>(x, qkv_w, qkv_b, qkv);
  attn_scores<<<dim3(HEADS, Bw), 256, 0, stream>>>(qkv, pos2, relb, rbs,
                                                   ri0, ri1, rwv, attn);
  transpose_k<<<dim3((P + 31) / 32, (Bw + 31) / 32), 256, 0, stream>>>(attn, attnT, Bw, P);
  topk_softmax<<<dim3(P), 256, 0, stream>>>(attnT, bias_p, ktop, Bw);
  transpose_k<<<dim3((Bw + 31) / 32, (P + 31) / 32), 256, 0, stream>>>(attnT, attn, P, Bw);
  av_gemm<<<dim3(HEADS, Bw), 256, 0, stream>>>(attn, qkv, mid);
  proj_gemm<<<dim3(DIM / 64, Bw), 256, 0, stream>>>(mid, proj_w, proj_b, out);
}
